// SingleScalePUMP_43954695307633
// MI455X (gfx1250) — matrix operand order repaired and checked by reading, not compiled, not run on hardware
//
#include <hip/hip_runtime.h>
#include <math.h>

typedef _Float16 v16h __attribute__((ext_vector_type(16)));
typedef _Float16 v8h  __attribute__((ext_vector_type(8)));
typedef float    v8f  __attribute__((ext_vector_type(8)));

#define PUMP_EPS   1e-8f
#define PUMP_NLPOW 1.4f

// ---------------------------------------------------------------------------
// Packing kernels
// ---------------------------------------------------------------------------

// desc2t[u][v][c] f16, u in [0,132) (= image row u-2), v in [0,148) (cols
// padded wide so the last 16-wide x-tile + dx never reads out of bounds).
__global__ void pump_pack_b(const float* __restrict__ desc2, _Float16* __restrict__ Bt) {
  const size_t total = (size_t)132 * 148 * 128;
  for (size_t i = blockIdx.x * (size_t)blockDim.x + threadIdx.x; i < total;
       i += (size_t)gridDim.x * blockDim.x) {
    int c = (int)(i % 128);
    size_t t = i / 128;
    int v = (int)(t % 148);
    int u = (int)(t / 148);
    int y = u - 2, x = v - 2;
    float val = 0.f;
    if (y >= 0 && y < 128 && x >= 0 && x < 128)
      val = desc2[(size_t)c * 16384 + (size_t)y * 128 + x];
    Bt[i] = (_Float16)val;
  }
}

// Apack[(dy*4+dx)][k][c] f16 = desc1[c, 4*pi+dy, 4*pj+dx], k = pi*32+pj
__global__ void pump_pack_a(const float* __restrict__ desc1, _Float16* __restrict__ Apack) {
  const size_t total = (size_t)16 * 1024 * 128;
  for (size_t i = blockIdx.x * (size_t)blockDim.x + threadIdx.x; i < total;
       i += (size_t)gridDim.x * blockDim.x) {
    int c = (int)(i % 128);
    size_t t = i / 128;
    int k = (int)(t % 1024);
    int d = (int)(t / 1024);
    int dy = d >> 2, dx = d & 3;
    int pi = k >> 5, pj = k & 31;
    Apack[i] = (_Float16)desc1[(size_t)c * 16384 + (size_t)(4 * pi + dy) * 128 + (4 * pj + dx)];
  }
}

// pnorm[k] = ||patch_k||2 (from f32 desc1), weights0[k] = pnorm>0
__global__ void pump_pnorm(const float* __restrict__ desc1, float* __restrict__ pnorm,
                           float* __restrict__ w0) {
  int k = blockIdx.x * (blockDim.x >> 5) + (threadIdx.x >> 5);
  if (k >= 1024) return;
  int lane = threadIdx.x & 31;
  int pi = k >> 5, pj = k & 31;
  float s = 0.f;
  for (int e = lane; e < 2048; e += 32) {
    int c = e >> 4, dy = (e >> 2) & 3, dx = e & 3;
    float v = desc1[(size_t)c * 16384 + (size_t)(4 * pi + dy) * 128 + (4 * pj + dx)];
    s += v * v;
  }
#pragma unroll
  for (int m = 16; m >= 1; m >>= 1) s += __shfl_xor(s, m, 32);
  if (lane == 0) {
    pnorm[k] = sqrtf(s);
    w0[k] = (s > 0.f) ? 1.f : 0.f;
  }
}

__global__ void pump_sq(const float* __restrict__ desc2, float* __restrict__ sq) {
  int i = blockIdx.x * blockDim.x + threadIdx.x;
  if (i >= 16384) return;
  float s = 0.f;
  for (int c = 0; c < 128; ++c) {
    float v = desc2[(size_t)c * 16384 + i];
    s += v * v;
  }
  sq[i] = s;
}

__global__ void pump_wnorm(const float* __restrict__ sq, float* __restrict__ wn) {
  int i = blockIdx.x * blockDim.x + threadIdx.x;
  if (i >= 129 * 129) return;
  int y = i / 129, x = i % 129;
  float s = 0.f;
  for (int dy = 0; dy < 4; ++dy) {
    int r = y - 2 + dy;
    if (r < 0 || r >= 128) continue;
    for (int dx = 0; dx < 4; ++dx) {
      int c = x - 2 + dx;
      if (c < 0 || c >= 128) continue;
      s += sq[r * 128 + c];
    }
  }
  wn[i] = sqrtf(s);
}

// ---------------------------------------------------------------------------
// WMMA correlation GEMM: M=1024 patches, N=129x129 positions, K=2048
// One wave -> 64(M) x 16(N) tile for one y row. 256 v_wmma per wave.
// ---------------------------------------------------------------------------

__device__ __forceinline__ v16h pump_load_a16(const _Float16* __restrict__ p) {
  // A lane fragment: contiguous K chunk [0..16) -> VGPRs 0-7
  v8h lo = *(const v8h*)p;
  v8h hi = *(const v8h*)(p + 8);
  v16h f;
#pragma unroll
  for (int i = 0; i < 8; ++i) { f[i] = lo[i]; f[i + 8] = hi[i]; }
  return f;
}

__global__ void __launch_bounds__(256) pump_corr_wmma(
    const _Float16* __restrict__ Apack, const _Float16* __restrict__ Bt,
    const float* __restrict__ pnorm, const float* __restrict__ wnorm,
    float* __restrict__ pyr0) {
  int wid = blockIdx.x * 8 + (threadIdx.x >> 5);  // 18576 waves exactly
  int lane = threadIdx.x & 31;
  int n = lane & 15, hi = lane >> 4;
  int mb = wid & 15;          // 16 blocks of 64 patch rows
  int xt = (wid >> 4) % 9;    // 9 x-tiles of 16
  int y = wid / 144;          // 129 rows
  int x0 = xt * 16;

  v8f acc0 = {}, acc1 = {}, acc2 = {}, acc3 = {};
  for (int d = 0; d < 16; ++d) {  // (dy,dx) window offsets
    int dy = d >> 2, dx = d & 3;
    const _Float16* bb = Bt + ((size_t)((y + dy) * 148 + x0 + n + dx)) * 128;
    const _Float16* ab = Apack + ((size_t)d * 1024 + (size_t)mb * 64 + n) * 128;
#pragma unroll
    for (int cb = 0; cb < 4; ++cb) {  // K=128 channels in 4 steps of 32
      int co = cb * 32;
      // B lane fragment: contiguous K 0-15 (lanes 0-15) / 16-31 (lanes 16-31)
      v16h bf = *(const v16h*)(bb + co + hi * 16);
      v16h a0 = pump_load_a16(ab + co + hi * 16);
      v16h a1 = pump_load_a16(ab + 16 * 128 + co + hi * 16);
      v16h a2 = pump_load_a16(ab + 32 * 128 + co + hi * 16);
      v16h a3 = pump_load_a16(ab + 48 * 128 + co + hi * 16);
      acc0 = __builtin_amdgcn_wmma_f32_16x16x32_f16(false, a0, false, bf, (short)0, acc0, false, false);
      acc1 = __builtin_amdgcn_wmma_f32_16x16x32_f16(false, a1, false, bf, (short)0, acc1, false, false);
      acc2 = __builtin_amdgcn_wmma_f32_16x16x32_f16(false, a2, false, bf, (short)0, acc2, false, false);
      acc3 = __builtin_amdgcn_wmma_f32_16x16x32_f16(false, a3, false, bf, (short)0, acc3, false, false);
    }
  }

  int x = x0 + n;
  if (x < 129) {
    float wn = wnorm[y * 129 + x];
    v8f accs[4] = {acc0, acc1, acc2, acc3};
#pragma unroll
    for (int t = 0; t < 4; ++t) {
#pragma unroll
      for (int r = 0; r < 8; ++r) {
        int row = mb * 64 + t * 16 + hi * 8 + r;  // acc VGPR r: M=r (lo) / r+8 (hi)
        float c = accs[t][r] / (pnorm[row] * wn + PUMP_EPS);
        c = fmaxf(c, 0.f);
        pyr0[(size_t)row * 16641 + (size_t)y * 129 + x] = powf(c, PUMP_NLPOW);
      }
    }
  }
}

// ---------------------------------------------------------------------------
// Forward pyramid: fused maxpool(3x3,s2,p1) + sparse_conv_fwd + activation
// ---------------------------------------------------------------------------

__global__ void pump_sparse_fwd(const float* __restrict__ lo, const float* __restrict__ w,
                                float* __restrict__ up, int Hb, int Wb, int h, int wd,
                                int Hu, int Wu, int hu, int wu) {
  size_t total = (size_t)Hu * Wu * hu * wu;
  for (size_t idx = blockIdx.x * (size_t)blockDim.x + threadIdx.x; idx < total;
       idx += (size_t)gridDim.x * blockDim.x) {
    int x = (int)(idx % wu);
    size_t t = idx / wu;
    int y = (int)(t % hu);
    t /= hu;
    int J = (int)(t % Wu);
    int I = (int)(t / Wu);
    float num = 0.f, den = 0.f;
#pragma unroll
    for (int di = 0; di < 2; ++di)
#pragma unroll
      for (int dj = 0; dj < 2; ++dj) {
        int bi = 2 * I + di, bj = 2 * J + dj;
        if (bi >= Hb || bj >= Wb) continue;  // implicit zero-pad (odd dims)
        float wv = w[bi * Wb + bj];
        den += wv;
        int yy = y - (2 * di - 1), xx = x - (2 * dj - 1);  // shift2 by (2di-1,2dj-1)
        if (yy < 0 || yy >= hu || xx < 0 || xx >= wu) continue;
        // pooled[bi,bj,yy,xx] = 3x3 max of lo centered at (2yy,2xx)
        const float* lp = lo + ((size_t)bi * Wb + bj) * (size_t)h * wd;
        float mx = -INFINITY;
        for (int u = -1; u <= 1; ++u) {
          int r = 2 * yy + u;
          if (r < 0 || r >= h) continue;
          for (int v = -1; v <= 1; ++v) {
            int cc = 2 * xx + v;
            if (cc < 0 || cc >= wd) continue;
            mx = fmaxf(mx, lp[(size_t)r * wd + cc]);
          }
        }
        num += wv * mx;
      }
    float o = num / fmaxf(den, 3.6f);  // 4*BORDER_INV
    o = fmaxf(o, 0.f);
    up[idx] = powf(o, PUMP_NLPOW);
  }
}

__global__ void pump_weights(const float* __restrict__ w, float* __restrict__ wn,
                             int Hb, int Wb, int Hu, int Wu) {
  int idx = blockIdx.x * blockDim.x + threadIdx.x;
  if (idx >= Hu * Wu) return;
  int J = idx % Wu, I = idx / Wu;
  float s = 0.f;
  for (int di = 0; di < 2; ++di)
    for (int dj = 0; dj < 2; ++dj) {
      int bi = 2 * I + di, bj = 2 * J + dj;
      if (bi < Hb && bj < Wb) s += w[bi * Wb + bj];
    }
  wn[idx] = 0.25f * s;
}

// ---------------------------------------------------------------------------
// Backward: fused sparse_conv_rev + max_unpool (maxpool recomputed on the fly)
// ---------------------------------------------------------------------------

__global__ void pump_unpool(const float* __restrict__ up, const float* __restrict__ lo,
                            float* __restrict__ out, int Hb, int Wb, int h, int wd,
                            int Hu, int Wu, int hu, int wu) {
  size_t total = (size_t)Hb * Wb * h * wd;
  for (size_t idx = blockIdx.x * (size_t)blockDim.x + threadIdx.x; idx < total;
       idx += (size_t)gridDim.x * blockDim.x) {
    int x = (int)(idx % wd);
    size_t t = idx / wd;
    int y = (int)(t % h);
    t /= h;
    int bj = (int)(t % Wb);
    int bi = (int)(t / Wb);
    int hm = (h - 1) / 2 + 1, wm = (wd - 1) / 2 + 1;  // == hu, wu
    int a = min(y >> 1, hm - 1), b = min(x >> 1, wm - 1);
    // up_mp = mp[bi,bj,a,b] : 3x3 max of lo centered (2a,2b)
    const float* lp = lo + ((size_t)bi * Wb + bj) * (size_t)h * wd;
    float mx = -INFINITY;
    for (int u = -1; u <= 1; ++u) {
      int r = 2 * a + u;
      if (r < 0 || r >= h) continue;
      for (int v = -1; v <= 1; ++v) {
        int cc = 2 * b + v;
        if (cc < 0 || cc >= wd) continue;
        mx = fmaxf(mx, lp[(size_t)r * wd + cc]);
      }
    }
    // up_val = sparse_conv_rev(upper)[bi,bj,a,b] = upper[bi/2,bj/2,a-1+2di,b-1+2dj]
    int ui = bi >> 1, uj = bj >> 1;
    int ry = a - 1 + 2 * (bi & 1), rx = b - 1 + 2 * (bj & 1);
    float uv = 0.f;
    if (ui < Hu && uj < Wu && ry >= 0 && ry < hu && rx >= 0 && rx < wu)
      uv = up[(((size_t)ui * Wu + uj) * hu + ry) * wu + rx];
    float lov = lp[(size_t)y * wd + x];
    out[idx] = (lov == mx) ? fmaxf(lov, uv) : lov;
  }
}

// ---------------------------------------------------------------------------
// Host orchestration
// ---------------------------------------------------------------------------

static inline int pump_nblk(size_t n) { return (int)((n + 255) / 256); }

extern "C" void kernel_launch(void* const* d_in, const int* in_sizes, int n_in,
                              void* d_out, int out_size, void* d_ws, size_t ws_size,
                              hipStream_t stream) {
  (void)in_sizes; (void)n_in; (void)out_size; (void)ws_size;
  const float* desc1 = (const float*)d_in[0];
  const float* desc2 = (const float*)d_in[1];
  float* out = (float*)d_out;
  char* ws = (char*)d_ws;

  // deterministic bump allocator over workspace
  size_t cur = 0;
  auto alloc = [&](size_t bytes) -> size_t {
    size_t o = cur;
    cur += (bytes + 255) & ~(size_t)255;
    return o;
  };

  const size_t oBt = alloc((size_t)132 * 148 * 128 * 2);   // 5.0 MB (reused in backward)
  const size_t oA  = alloc((size_t)16 * 1024 * 128 * 2);   // 4.2 MB
  const size_t oPn = alloc(1024 * 4);
  const size_t oSq = alloc(16384 * 4);
  const size_t oWn = alloc(129 * 129 * 4);

  _Float16* Bt = (_Float16*)(ws + oBt);
  _Float16* Ap = (_Float16*)(ws + oA);
  float* pn = (float*)(ws + oPn);
  float* sq = (float*)(ws + oSq);
  float* wn = (float*)(ws + oWn);

  // ---- first_level prep + WMMA GEMM ----
  pump_pack_b<<<pump_nblk((size_t)132 * 148 * 128), 256, 0, stream>>>(desc2, Bt);
  pump_pack_a<<<pump_nblk((size_t)16 * 1024 * 128), 256, 0, stream>>>(desc1, Ap);
  pump_sq<<<pump_nblk(16384), 256, 0, stream>>>(desc2, sq);
  pump_wnorm<<<pump_nblk(129 * 129), 256, 0, stream>>>(sq, wn);

  struct Lv { int Hb, Wb, h, w; float* p; float* wt; };
  Lv L[10];
  L[0].Hb = 32; L[0].Wb = 32; L[0].h = 129; L[0].w = 129;
  L[0].p = (float*)(ws + alloc((size_t)32 * 32 * 129 * 129 * 4));  // 68 MB
  L[0].wt = (float*)(ws + alloc(32 * 32 * 4));

  pump_pnorm<<<128, 256, 0, stream>>>(desc1, pn, L[0].wt);
  // 18576 waves = 2322 blocks of 8 waves (exact)
  pump_corr_wmma<<<2322, 256, 0, stream>>>(Ap, Bt, pn, wn, L[0].p);

  // ---- forward pyramid ----
  int nl = 1;
  for (int i = 1; i <= 9; ++i) {
    Lv& P = L[i - 1];
    int Hu = (P.Hb + 1) / 2, Wu = (P.Wb + 1) / 2;
    int hu = (P.h - 1) / 2 + 1, wu = (P.w - 1) / 2 + 1;
    L[i].Hb = Hu; L[i].Wb = Wu; L[i].h = hu; L[i].w = wu;
    L[i].p = (float*)(ws + alloc((size_t)Hu * Wu * hu * wu * 4));
    L[i].wt = (float*)(ws + alloc((size_t)Hu * Wu * 4));
    size_t total = (size_t)Hu * Wu * hu * wu;
    pump_sparse_fwd<<<pump_nblk(total), 256, 0, stream>>>(P.p, P.wt, L[i].p,
        P.Hb, P.Wb, P.h, P.w, Hu, Wu, hu, wu);
    pump_weights<<<pump_nblk((size_t)Hu * Wu), 256, 0, stream>>>(P.wt, L[i].wt,
        P.Hb, P.Wb, Hu, Wu);
    ++nl;
    if (hu <= 5 || wu <= 5) break;  // min(shape) <= MIN_SHAPE
  }

  // ---- backward pyramid (overlay scratch on dead desc2t region) ----
  size_t bcur = oBt;
  auto balloc = [&](size_t bytes) -> size_t {
    size_t o = bcur;
    bcur += (bytes + 255) & ~(size_t)255;
    return o;
  };
  for (int i = nl - 1; i >= 1; --i) {
    Lv& U = L[i];
    Lv& Lo = L[i - 1];
    size_t total = (size_t)Lo.Hb * Lo.Wb * Lo.h * Lo.w;
    float* dst = (i == 1) ? out : (float*)(ws + balloc(total * 4));
    pump_unpool<<<pump_nblk(total), 256, 0, stream>>>(U.p, Lo.p, dst,
        Lo.Hb, Lo.Wb, Lo.h, Lo.w, U.Hb, U.Wb, U.h, U.w);
    L[i - 1].p = dst;  // becomes "upper" for the next (lower) iteration
  }
}
